// EncoderBlock_37598143709652
// MI455X (gfx1250) — matrix operand order repaired and checked by reading, not compiled, not run on hardware
//
#include <hip/hip_runtime.h>
#include <math.h>

// ---------- types ----------
typedef __attribute__((ext_vector_type(16))) __bf16        v16bf;
typedef __attribute__((ext_vector_type(16))) unsigned short v16us;
typedef __attribute__((ext_vector_type(8)))  float          v8f;

// ---------- bf16 helpers (manual, storage = unsigned short) ----------
__device__ __forceinline__ float bf2f(unsigned short u) {
    unsigned int v = ((unsigned int)u) << 16;
    float f; __builtin_memcpy(&f, &v, 4); return f;
}
__device__ __forceinline__ unsigned short f2bf(float f) {
    unsigned int u; __builtin_memcpy(&u, &f, 4);
    unsigned int r = u + 0x7FFFu + ((u >> 16) & 1u);   // round-to-nearest-even
    return (unsigned short)(r >> 16);
}

// ---------- DPP16 cross-lane reductions within each 16-lane row (wave32) ----------
// quad_perm xor1 = 0xB1, quad_perm xor2 = 0x4E, row_half_mirror = 0x141, row_mirror = 0x140
template <int CTRL>
__device__ __forceinline__ float dpp_movf(float x) {
    int xi = __builtin_bit_cast(int, x);
    int r  = __builtin_amdgcn_update_dpp(0, xi, CTRL, 0xF, 0xF, true);
    return __builtin_bit_cast(float, r);
}
__device__ __forceinline__ float row16_max(float x) {
    x = fmaxf(x, dpp_movf<0xB1>(x));    // xor 1 (quad)
    x = fmaxf(x, dpp_movf<0x4E>(x));    // xor 2 (quad)
    x = fmaxf(x, dpp_movf<0x141>(x));   // half mirror (== xor4 on quad-uniform data)
    x = fmaxf(x, dpp_movf<0x140>(x));   // row mirror  (== xor8 on octet-uniform data)
    return x;
}
__device__ __forceinline__ float row16_sum(float x) {
    x += dpp_movf<0xB1>(x);
    x += dpp_movf<0x4E>(x);
    x += dpp_movf<0x141>(x);
    x += dpp_movf<0x140>(x);
    return x;
}

// ---------- WMMA wrapper: D = A(16x32 bf16) * B(32x16 bf16) + C(16x16 f32) ----------
__device__ __forceinline__ v8f wmma_bf16(v16us a, v16us b, v8f c) {
    union U { v16us u; v16bf b; };
    U ua; ua.u = a;
    U ub; ub.u = b;
    return __builtin_amdgcn_wmma_f32_16x16x32_bf16(
        /*neg_a=*/false, ua.b, /*neg_b=*/false, ub.b,
        /*c_mod=*/(short)0, c, /*reuse_a=*/false, /*reuse_b=*/false);
}

// A-fragment: 16x32 bf16 row-major at `base`, leading dim `ld` (elements).
// lanes 0-15 -> M=lane, K=0..7 & 16..23 ; lanes 16-31 -> M=lane-16, K=8..15 & 24..31
__device__ __forceinline__ v16us load_frag_a(const unsigned short* base, int ld) {
    int lane = threadIdx.x & 31;
    int row  = lane & 15;
    int koff = (lane >> 4) << 3;           // 0 or 8
    const unsigned short* p = base + row * ld + koff;
    union { uint4 q[2]; v16us v; } t;
    t.q[0] = *reinterpret_cast<const uint4*>(p);        // K = koff .. koff+7
    t.q[1] = *reinterpret_cast<const uint4*>(p + 16);   // K = 16+koff .. 16+koff+7
    return t.v;
}

// B-fragment from K-contiguous storage: element (kk, n) = base[n*ld + kk], K=32, N=16.
// Per-lane data is two runs of 8 shorts (kk=kbase+0..7, kbase+16..23) -> two uint4 LDS loads.
__device__ __forceinline__ v16us load_frag_bt(const unsigned short* base, int ld) {
    int lane  = threadIdx.x & 31;
    int col   = lane & 15;
    int kbase = (lane >> 4) << 3;          // 0 or 8
    const unsigned short* p = base + col * ld + kbase;
    union { uint4 q[2]; v16us v; } t;
    t.q[0] = *reinterpret_cast<const uint4*>(p);
    t.q[1] = *reinterpret_cast<const uint4*>(p + 16);
    return t.v;
}

// ---------- fp32 -> bf16 cast (weights) ----------
__global__ __launch_bounds__(256) void cast_f32_bf16(const float* __restrict__ in,
                                                     unsigned short* __restrict__ out, int n4) {
    int i = blockIdx.x * 256 + threadIdx.x;
    if (i < n4) {
        float4 v = reinterpret_cast<const float4*>(in)[i];
        uint2 o;
        o.x = (unsigned int)f2bf(v.x) | ((unsigned int)f2bf(v.y) << 16);
        o.y = (unsigned int)f2bf(v.z) | ((unsigned int)f2bf(v.w) << 16);
        reinterpret_cast<uint2*>(out)[i] = o;
    }
}

// ---------- layernorm: fp32 in (row of D=1024), bf16 out ----------
__global__ __launch_bounds__(256) void layernorm_bf16(const float* __restrict__ x,
                                                      const float* __restrict__ g,
                                                      const float* __restrict__ bta,
                                                      unsigned short* __restrict__ out) {
    const int Dn = 1024;
    int row = blockIdx.x;
    const float* xr = x + (size_t)row * Dn;
    float4 v = reinterpret_cast<const float4*>(xr)[threadIdx.x];
    float s  = v.x + v.y + v.z + v.w;
    float ss = v.x*v.x + v.y*v.y + v.z*v.z + v.w*v.w;
    // intra-row (16-lane) via DPP, then one cross-row shuffle
    s = row16_sum(s);  ss = row16_sum(ss);
    s += __shfl_xor(s, 16, 32);
    ss += __shfl_xor(ss, 16, 32);
    __shared__ float rs[8], rss[8];
    int wave = threadIdx.x >> 5, lane = threadIdx.x & 31;
    if (lane == 0) { rs[wave] = s; rss[wave] = ss; }
    __syncthreads();
    if (wave == 0) {
        float a = (lane < 8) ? rs[lane] : 0.0f;
        float b = (lane < 8) ? rss[lane] : 0.0f;
        #pragma unroll
        for (int m = 1; m < 8; m <<= 1) { a += __shfl_xor(a, m, 32); b += __shfl_xor(b, m, 32); }
        if (lane == 0) { rs[0] = a; rss[0] = b; }
    }
    __syncthreads();
    float mean = rs[0] * (1.0f / Dn);
    float var  = rss[0] * (1.0f / Dn) - mean * mean;
    float rstd = rsqrtf(var + 1e-5f);
    int c0 = threadIdx.x * 4;
    float xs[4] = { v.x, v.y, v.z, v.w };
    #pragma unroll
    for (int u = 0; u < 4; ++u) {
        int c = c0 + u;
        out[(size_t)row * Dn + c] = f2bf((xs[u] - mean) * rstd * g[c] + bta[c]);
    }
}

// ---------- generic bf16 GEMM, double-buffered LDS: C = A(MxK)*B(KxN)+bias ----------
// EPI 0: +bias            -> bf16 out
// EPI 1: +bias, exact GELU-> bf16 out
// EPI 2: +bias, +residual -> f32 out
template <int EPI>
__global__ __launch_bounds__(256) void gemm_bf16(const unsigned short* __restrict__ A,
                                                 const unsigned short* __restrict__ Bw,
                                                 const float* __restrict__ bias,
                                                 const float* __restrict__ res,
                                                 float* __restrict__ Cf,
                                                 unsigned short* __restrict__ Cb,
                                                 int M, int N, int K) {
    __shared__ __align__(16) unsigned short sA[2][128 * 32];   // [buf][row][kk]
    __shared__ __align__(16) unsigned short sBt[2][128 * 32];  // [buf][n][kk]
    const int tile_n = blockIdx.x * 128;
    const int tile_m = blockIdx.y * 128;
    const int tid  = threadIdx.x;
    const int wave = tid >> 5;
    const int wm = (wave >> 1) * 32;     // 0,32,64,96
    const int wn = (wave & 1) * 64;      // 0,64
    const int lane  = tid & 31;
    const int colL  = lane & 15;
    const int rowbL = (lane >> 4) * 8;

    v8f acc[2][4];
    #pragma unroll
    for (int i = 0; i < 2; ++i)
        #pragma unroll
        for (int j = 0; j < 4; ++j) acc[i][j] = {};

    const int arow = tid >> 1;            // 0..127
    const int acol = (tid & 1) * 16;      // 0 / 16
    const int brow = tid >> 3;            // 0..31
    const int bseg = (tid & 7) * 16;      // 0..112

    const unsigned short* gAp = A  + (size_t)(tile_m + arow) * K + acol;
    const unsigned short* gBp = Bw + (size_t)brow * N + tile_n + bseg;

    auto gload = [&](int k0, uint4& A0, uint4& A1, uint4& B0, uint4& B1) {
        const uint4* ga = reinterpret_cast<const uint4*>(gAp + k0);
        A0 = ga[0]; A1 = ga[1];
        const uint4* gb = reinterpret_cast<const uint4*>(gBp + (size_t)k0 * N);
        B0 = gb[0]; B1 = gb[1];
    };
    auto lstore = [&](int buf, uint4 A0, uint4 A1, uint4 B0, uint4 B1) {
        *reinterpret_cast<uint4*>(&sA[buf][arow * 32 + acol])     = A0;
        *reinterpret_cast<uint4*>(&sA[buf][arow * 32 + acol + 8]) = A1;
        union { uint4 q[2]; unsigned short s[16]; } t;
        t.q[0] = B0; t.q[1] = B1;
        #pragma unroll
        for (int u = 0; u < 16; ++u) sBt[buf][(bseg + u) * 32 + brow] = t.s[u];
    };

    const int nk = K >> 5;
    {   // prologue: stage chunk 0
        uint4 A0, A1, B0, B1;
        gload(0, A0, A1, B0, B1);
        lstore(0, A0, A1, B0, B1);
    }
    __syncthreads();

    for (int ik = 0; ik < nk; ++ik) {
        const int buf = ik & 1;
        uint4 nA0 = {}, nA1 = {}, nB0 = {}, nB1 = {};
        if (ik + 1 < nk) gload((ik + 1) << 5, nA0, nA1, nB0, nB1);
        if (ik + 2 < nk) {   // warm GL2 for the chunk after next
            __builtin_prefetch(gAp + ((ik + 2) << 5), 0, 1);
            __builtin_prefetch(gBp + ((size_t)((ik + 2) << 5)) * N, 0, 1);
        }
        v16us af0 = load_frag_a(&sA[buf][(wm + 0)  * 32], 32);
        v16us af1 = load_frag_a(&sA[buf][(wm + 16) * 32], 32);
        #pragma unroll
        for (int j = 0; j < 4; ++j) {
            v16us bf = load_frag_bt(&sBt[buf][(wn + 16 * j) * 32], 32);
            acc[0][j] = wmma_bf16(af0, bf, acc[0][j]);
            acc[1][j] = wmma_bf16(af1, bf, acc[1][j]);
        }
        if (ik + 1 < nk) lstore(buf ^ 1, nA0, nA1, nB0, nB1);
        __syncthreads();
    }

    // epilogue
    #pragma unroll
    for (int i = 0; i < 2; ++i) {
        #pragma unroll
        for (int j = 0; j < 4; ++j) {
            int gr0 = tile_m + wm + 16 * i + rowbL;
            int gc  = tile_n + wn + 16 * j + colL;
            float bv = bias[gc];
            #pragma unroll
            for (int r = 0; r < 8; ++r) {
                float vv = acc[i][j][r] + bv;
                size_t idx = (size_t)(gr0 + r) * N + gc;
                if (EPI == 1) {
                    vv = 0.5f * vv * (1.0f + erff(vv * 0.70710678118654752f));
                }
                if (EPI == 2) {
                    vv += res[idx];
                    Cf[idx] = vv;
                } else {
                    Cb[idx] = f2bf(vv);
                }
            }
        }
    }
}

// ---------- flash attention: q,k,v bf16 (B*S x D, head-strided), out bf16 ----------
__global__ __launch_bounds__(128) void attention_kernel(const unsigned short* __restrict__ q,
                                                        const unsigned short* __restrict__ k,
                                                        const unsigned short* __restrict__ v,
                                                        unsigned short* __restrict__ o) {
    const int S = 1024, Dd = 1024, DH = 64;
    __shared__ __align__(16) unsigned short sk[2][32 * 64];     // [buf][key][dh]
    __shared__ __align__(16) unsigned short svt[2][64 * 32];    // [buf][dh][key]
    __shared__ __align__(16) unsigned short sp[4][16 * 32];     // per-wave P tile

    const int bh = blockIdx.y;
    const int bb = bh >> 4;          // / H (H=16)
    const int hh = bh & 15;
    const int wave = threadIdx.x >> 5;
    const int lane = threadIdx.x & 31;
    const int colL = lane & 15;
    const int rowbL = (lane >> 4) * 8;

    const size_t baseRow = (size_t)bb * S;
    const unsigned short* qb = q + baseRow * Dd + hh * DH;
    const unsigned short* kb = k + baseRow * Dd + hh * DH;
    const unsigned short* vb = v + baseRow * Dd + hh * DH;
    const int qrow0 = blockIdx.x * 64 + wave * 16;

    v16us a0 = load_frag_a(qb + (size_t)qrow0 * Dd, Dd);        // dh 0..31
    v16us a1 = load_frag_a(qb + (size_t)qrow0 * Dd + 32, Dd);   // dh 32..63

    v8f o0 = {}, o1 = {}, o2 = {}, o3 = {};
    float mst[8], lst[8];
    #pragma unroll
    for (int r = 0; r < 8; ++r) { mst[r] = -3.0e38f; lst[r] = 0.0f; }
    const float rscale = 0.03125f;   // 1/sqrt(D=1024)

    const int key = threadIdx.x >> 2;        // 0..31
    const int seg = (threadIdx.x & 3) * 16;  // dh segment

    auto kv_gload = [&](int kc, uint4& K0, uint4& K1, uint4& V0, uint4& V1) {
        const uint4* gk = reinterpret_cast<const uint4*>(kb + (size_t)(kc + key) * Dd + seg);
        K0 = gk[0]; K1 = gk[1];
        const uint4* gv = reinterpret_cast<const uint4*>(vb + (size_t)(kc + key) * Dd + seg);
        V0 = gv[0]; V1 = gv[1];
    };
    auto kv_store = [&](int buf, uint4 K0, uint4 K1, uint4 V0, uint4 V1) {
        *reinterpret_cast<uint4*>(&sk[buf][key * 64 + seg])     = K0;
        *reinterpret_cast<uint4*>(&sk[buf][key * 64 + seg + 8]) = K1;
        union { uint4 q2[2]; unsigned short s[16]; } t;
        t.q2[0] = V0; t.q2[1] = V1;
        #pragma unroll
        for (int u = 0; u < 16; ++u) svt[buf][(seg + u) * 32 + key] = t.s[u];
    };

    {   // prologue: stage chunk 0
        uint4 K0, K1, V0, V1;
        kv_gload(0, K0, K1, V0, V1);
        kv_store(0, K0, K1, V0, V1);
    }
    __syncthreads();

    const int nkc = S >> 5;
    for (int ik = 0; ik < nkc; ++ik) {
        const int buf = ik & 1;
        uint4 nK0 = {}, nK1 = {}, nV0 = {}, nV1 = {};
        if (ik + 1 < nkc) kv_gload((ik + 1) << 5, nK0, nK1, nV0, nV1);

        // scores: S_tile(16x32) = Q(16x64) . K^T ; element(kk=dh, n=key) = sk[key*64+dh]
        const unsigned short* skb = sk[buf];
        v8f s0 = {}, s1 = {};
        s0 = wmma_bf16(a0, load_frag_bt(skb, 64), s0);
        s0 = wmma_bf16(a1, load_frag_bt(skb + 32, 64), s0);
        s1 = wmma_bf16(a0, load_frag_bt(skb + 16 * 64, 64), s1);
        s1 = wmma_bf16(a1, load_frag_bt(skb + 16 * 64 + 32, 64), s1);

        // online softmax per row; reductions stay inside each 16-lane DPP row
        #pragma unroll
        for (int r = 0; r < 8; ++r) {
            float a = s0[r] * rscale;
            float b = s1[r] * rscale;
            float mx = row16_max(fmaxf(a, b));
            float mnew = fmaxf(mst[r], mx);
            float corr = __expf(mst[r] - mnew);
            float p0 = __expf(a - mnew);
            float p1 = __expf(b - mnew);
            float ls = row16_sum(p0 + p1);
            lst[r] = lst[r] * corr + ls;
            mst[r] = mnew;
            o0[r] *= corr; o1[r] *= corr; o2[r] *= corr; o3[r] *= corr;
            s0[r] = p0; s1[r] = p1;
        }

        // re-layout P (C-frag) -> A-frag via LDS; same-wave LDS ops are in-order,
        // so no barrier is needed for this wave-private round-trip.
        unsigned short* spw = sp[wave];
        #pragma unroll
        for (int r = 0; r < 8; ++r) {
            spw[(rowbL + r) * 32 + colL]      = f2bf(s0[r]);
            spw[(rowbL + r) * 32 + 16 + colL] = f2bf(s1[r]);
        }
        v16us ap = load_frag_a(spw, 32);

        const unsigned short* svb = svt[buf];
        o0 = wmma_bf16(ap, load_frag_bt(svb + 0 * 16 * 32, 32), o0);
        o1 = wmma_bf16(ap, load_frag_bt(svb + 1 * 16 * 32, 32), o1);
        o2 = wmma_bf16(ap, load_frag_bt(svb + 2 * 16 * 32, 32), o2);
        o3 = wmma_bf16(ap, load_frag_bt(svb + 3 * 16 * 32, 32), o3);

        if (ik + 1 < nkc) kv_store(buf ^ 1, nK0, nK1, nV0, nV1);
        __syncthreads();
    }

    // normalize + write (merged-head layout: col = h*64 + dh)
    #pragma unroll
    for (int r = 0; r < 8; ++r) {
        size_t rowg = baseRow + qrow0 + rowbL + r;
        float inv = 1.0f / lst[r];
        o[rowg * Dd + hh * DH + 0  + colL] = f2bf(o0[r] * inv);
        o[rowg * Dd + hh * DH + 16 + colL] = f2bf(o1[r] * inv);
        o[rowg * Dd + hh * DH + 32 + colL] = f2bf(o2[r] * inv);
        o[rowg * Dd + hh * DH + 48 + colL] = f2bf(o3[r] * inv);
    }
}

// =======================================================================
extern "C" void kernel_launch(void* const* d_in, const int* in_sizes, int n_in,
                              void* d_out, int out_size, void* d_ws, size_t ws_size,
                              hipStream_t stream) {
    (void)in_sizes; (void)n_in; (void)out_size; (void)ws_size;
    constexpr int Bb = 4, S = 1024, Dd = 1024, Hh = 16, FFN = 4096;
    constexpr int BS = Bb * S;

    const float* x    = (const float*)d_in[0];
    const float* wq   = (const float*)d_in[1];
    const float* bq   = (const float*)d_in[2];
    const float* wk   = (const float*)d_in[3];
    const float* bk   = (const float*)d_in[4];
    const float* wv   = (const float*)d_in[5];
    const float* bv   = (const float*)d_in[6];
    const float* wo   = (const float*)d_in[7];
    const float* bo   = (const float*)d_in[8];
    const float* g1   = (const float*)d_in[9];
    const float* b1   = (const float*)d_in[10];
    const float* g2   = (const float*)d_in[11];
    const float* b2   = (const float*)d_in[12];
    const float* wfc1 = (const float*)d_in[13];
    const float* bfc1 = (const float*)d_in[14];
    const float* wfc2 = (const float*)d_in[15];
    const float* bfc2 = (const float*)d_in[16];
    float* out = (float*)d_out;

    char* p = (char*)d_ws;
    auto take = [&](size_t bytes) { char* r = p; p += (bytes + 255) & ~(size_t)255; return (void*)r; };
    unsigned short* wqB  = (unsigned short*)take((size_t)Dd * Dd * 2);
    unsigned short* wkB  = (unsigned short*)take((size_t)Dd * Dd * 2);
    unsigned short* wvB  = (unsigned short*)take((size_t)Dd * Dd * 2);
    unsigned short* woB  = (unsigned short*)take((size_t)Dd * Dd * 2);
    unsigned short* wf1B = (unsigned short*)take((size_t)Dd * FFN * 2);
    unsigned short* wf2B = (unsigned short*)take((size_t)FFN * Dd * 2);
    unsigned short* h1   = (unsigned short*)take((size_t)BS * Dd * 2);
    unsigned short* qB   = (unsigned short*)take((size_t)BS * Dd * 2);
    unsigned short* kB   = (unsigned short*)take((size_t)BS * Dd * 2);
    unsigned short* vB   = (unsigned short*)take((size_t)BS * Dd * 2);
    unsigned short* oB   = (unsigned short*)take((size_t)BS * Dd * 2);
    float*          att  = (float*)take((size_t)BS * Dd * 4);
    unsigned short* h2   = (unsigned short*)take((size_t)BS * Dd * 2);
    unsigned short* ffB  = (unsigned short*)take((size_t)BS * FFN * 2);

    auto cast = [&](const float* src, unsigned short* dst, size_t n) {
        int n4 = (int)(n / 4);
        cast_f32_bf16<<<(n4 + 255) / 256, 256, 0, stream>>>(src, dst, n4);
    };
    cast(wq, wqB, (size_t)Dd * Dd);
    cast(wk, wkB, (size_t)Dd * Dd);
    cast(wv, wvB, (size_t)Dd * Dd);
    cast(wo, woB, (size_t)Dd * Dd);
    cast(wfc1, wf1B, (size_t)Dd * FFN);
    cast(wfc2, wf2B, (size_t)FFN * Dd);

    // LN1
    layernorm_bf16<<<BS, 256, 0, stream>>>(x, g1, b1, h1);

    // Q,K,V projections (M=4096, N=1024, K=1024)
    dim3 gQKV(Dd / 128, BS / 128);
    gemm_bf16<0><<<gQKV, 256, 0, stream>>>(h1, wqB, bq, nullptr, nullptr, qB, BS, Dd, Dd);
    gemm_bf16<0><<<gQKV, 256, 0, stream>>>(h1, wkB, bk, nullptr, nullptr, kB, BS, Dd, Dd);
    gemm_bf16<0><<<gQKV, 256, 0, stream>>>(h1, wvB, bv, nullptr, nullptr, vB, BS, Dd, Dd);

    // attention (flash, online softmax, double-buffered K/V)
    dim3 gAtt(S / 64, Bb * Hh);
    attention_kernel<<<gAtt, 128, 0, stream>>>(qB, kB, vB, oB);

    // out projection + residual(x) -> att (f32)
    gemm_bf16<2><<<gQKV, 256, 0, stream>>>(oB, woB, bo, x, att, nullptr, BS, Dd, Dd);

    // LN2
    layernorm_bf16<<<BS, 256, 0, stream>>>(att, g2, b2, h2);

    // FFN fc1 + GELU (M=4096, N=4096, K=1024)
    dim3 gF1(FFN / 128, BS / 128);
    gemm_bf16<1><<<gF1, 256, 0, stream>>>(h2, wf1B, bfc1, nullptr, nullptr, ffB, BS, FFN, Dd);

    // FFN fc2 + residual(att) -> d_out (M=4096, N=1024, K=4096)
    dim3 gF2(Dd / 128, BS / 128);
    gemm_bf16<2><<<gF2, 256, 0, stream>>>(ffB, wf2B, bfc2, att, out, nullptr, BS, Dd, FFN);
}
